// TemporalRenderCrossMAEShift_53068615910056
// MI455X (gfx1250) — matrix operand order repaired and checked by reading, not compiled, not run on hardware
//
#include <hip/hip_runtime.h>
#include <math.h>

// ---------------------------------------------------------------------------
// TemporalRenderCrossMAEShift — shifted-window 3D attention block, CDNA5 WMMA
// All GEMMs: f16 A/B, f32 accumulate via v_wmma_f32_16x16x32_f16 (wave32).
// One wave computes a 64x64 tile (4x4 WMMA, 1:1 wmma:b128-load per K-step).
// amdgpu_waves_per_eu(1) lifts the 128-VGPR cap so accumulators don't spill.
// Workspace requirement: ~471 MB in d_ws.
// ---------------------------------------------------------------------------

typedef _Float16 half_t;
typedef __attribute__((ext_vector_type(16))) _Float16 v16h;
typedef __attribute__((ext_vector_type(8)))  _Float16 v8h;
typedef __attribute__((ext_vector_type(8)))  float    v8f;

#define Bb     2
#define Dd     8
#define Hh     64
#define Wdim   64
#define Cc     256
#define HEADS  2
#define HDim   128
#define Nn     512        // tokens per window (D*8*8)
#define Pp     64         // windows per image
#define BPw    128        // B*P windows
#define Mtok   65536      // BPw*Nn rows
#define HIDd   1024
#define SHIFT  4

// GEMM kernels: 1 wave per block, full register budget (no spills).
#define GEMM_ATTRS __attribute__((amdgpu_flat_work_group_size(32, 32), amdgpu_waves_per_eu(1)))

// ---- WMMA fragment loaders (per ISA 7.12.2 wave32 layouts) ----------------

// A: 16x32 f16, source row-major with leading dim lda (halves).
__device__ __forceinline__ v16h frag_a(const half_t* __restrict__ base, int lda) {
  int lane = threadIdx.x & 31;
  const half_t* p = base + (size_t)(lane & 15) * lda + ((lane >> 4) << 3);
  v8h lo = *(const v8h*)p;          // K = kb+0..7    (VGPR 0..3)
  v8h hi = *(const v8h*)(p + 16);   // K = 16+kb+0..7 (VGPR 4..7)
  v16h r;
#pragma unroll
  for (int i = 0; i < 8; ++i) { r[i] = lo[i]; r[i + 8] = hi[i]; }
  return r;
}

// B: 32x16 f16, loaded from the TRANSPOSED operand (rows = output cols),
// row-major with leading dim ldb -> per-lane K = kb+0..7 and 16+kb+0..7 (same as A).
__device__ __forceinline__ v16h frag_bt(const half_t* __restrict__ base, int ldb) {
  int lane = threadIdx.x & 31;
  const half_t* p = base + (size_t)(lane & 15) * ldb + ((lane >> 4) << 3);
  v8h lo = *(const v8h*)p;
  v8h hi = *(const v8h*)(p + 16);
  v16h r;
#pragma unroll
  for (int i = 0; i < 8; ++i) { r[i] = lo[i]; r[i + 8] = hi[i]; }
  return r;
}

// One wave: 64x64 output tile, K-loop step 32. 16 WMMA : 16 b128 loads.
__device__ __forceinline__ void mma64(const half_t* __restrict__ A, int lda,
                                      const half_t* __restrict__ Bt, int ldb,
                                      int K, v8f (&acc)[4][4]) {
  for (int kk = 0; kk < K; kk += 32) {
    v16h a[4], b[4];
#pragma unroll
    for (int i = 0; i < 4; ++i) a[i] = frag_a(A + (size_t)(16 * i) * lda + kk, lda);
#pragma unroll
    for (int j = 0; j < 4; ++j) b[j] = frag_bt(Bt + (size_t)(16 * j) * ldb + kk, ldb);
#pragma unroll
    for (int i = 0; i < 4; ++i)
#pragma unroll
      for (int j = 0; j < 4; ++j)
        acc[i][j] = __builtin_amdgcn_wmma_f32_16x16x32_f16(
            false, a[i], false, b[j], (short)0, acc[i][j], false, false);
  }
}

// ---- elementwise / reduction kernels --------------------------------------

__global__ void k_cast_f16(const float* __restrict__ s, half_t* __restrict__ d, int n) {
  int i = blockIdx.x * blockDim.x + threadIdx.x;
  if (i < n) d[i] = (half_t)s[i];
}

// LayerNorm(norm3) + roll(-4,-4) + window partition -> t16 (BPw, Nn, Cc) f16
__global__ void k_ln_shift_window(const float* __restrict__ x,
                                  const float* __restrict__ g, const float* __restrict__ b,
                                  half_t* __restrict__ t16) {
  __shared__ float ssum[256], ssq[256];
  int tid = blockIdx.x;                 // token id = w*512 + n
  int c = threadIdx.x;
  int w = tid >> 9, n = tid & 511;
  int bb = w >> 6, p = w & 63;
  int hn = p >> 3, wn = p & 7;
  int d = n >> 6, wh = (n >> 3) & 7, ww = n & 7;
  int sh = ((hn << 3) + wh + SHIFT) & 63;   // roll by -SHIFT == gather from +SHIFT
  int sw = ((wn << 3) + ww + SHIFT) & 63;
  size_t row = ((size_t)((bb * Dd + d) * Hh + sh) * Wdim + sw) * Cc;
  float v = x[row + c];
  ssum[c] = v; ssq[c] = v * v; __syncthreads();
  for (int off = 128; off > 0; off >>= 1) {
    if (c < off) { ssum[c] += ssum[c + off]; ssq[c] += ssq[c + off]; }
    __syncthreads();
  }
  float mean = ssum[0] * (1.0f / 256.0f);
  float var  = ssq[0] * (1.0f / 256.0f) - mean * mean;
  float rstd = rsqrtf(var + 1e-5f);
  t16[(size_t)tid * Cc + c] = (half_t)((v - mean) * rstd * g[c] + b[c]);
}

// LayerNorm(norm4) over flat rows of x1 -> ln16 (Mtok, Cc) f16
__global__ void k_ln_rows(const float* __restrict__ xr,
                          const float* __restrict__ g, const float* __restrict__ b,
                          half_t* __restrict__ ln16) {
  __shared__ float ssum[256], ssq[256];
  size_t row = (size_t)blockIdx.x * Cc;
  int c = threadIdx.x;
  float v = xr[row + c];
  ssum[c] = v; ssq[c] = v * v; __syncthreads();
  for (int off = 128; off > 0; off >>= 1) {
    if (c < off) { ssum[c] += ssum[c + off]; ssq[c] += ssq[c + off]; }
    __syncthreads();
  }
  float mean = ssum[0] * (1.0f / 256.0f);
  float var  = ssq[0] * (1.0f / 256.0f) - mean * mean;
  float rstd = rsqrtf(var + 1e-5f);
  ln16[row + c] = (half_t)((v - mean) * rstd * g[c] + b[c]);
}

// Row softmax over 512 f16 logits, in place.
__global__ void k_softmax(half_t* __restrict__ s) {
  __shared__ float red[256];
  half_t* p = s + (size_t)blockIdx.x * Nn;
  int t = threadIdx.x;
  float a = (float)p[t], b = (float)p[t + 256];
  red[t] = fmaxf(a, b); __syncthreads();
  for (int off = 128; off > 0; off >>= 1) {
    if (t < off) red[t] = fmaxf(red[t], red[t + off]);
    __syncthreads();
  }
  float mx = red[0]; __syncthreads();
  float e0 = __expf(a - mx), e1 = __expf(b - mx);
  red[t] = e0 + e1; __syncthreads();
  for (int off = 128; off > 0; off >>= 1) {
    if (t < off) red[t] += red[t + off];
    __syncthreads();
  }
  float inv = 1.0f / red[0];
  p[t] = (half_t)(e0 * inv);
  p[t + 256] = (half_t)(e1 * inv);
}

// ---- GEMM kernels (1 wave per 64x64 tile) ---------------------------------

// QKV: t16 (Mtok x 256) @ qkv_w^T (768 x 256) + bias.
// Epilogue splits cols into Q (scaled by hd^-0.5), K, V; V stored transposed.
__global__ void GEMM_ATTRS k_gemm_qkv(const half_t* __restrict__ t16,
                                      const half_t* __restrict__ wq,
                                      const float* __restrict__ bias,
                                      half_t* __restrict__ q16, half_t* __restrict__ k16,
                                      half_t* __restrict__ vt16) {
  int m0 = blockIdx.x * 64, o0 = blockIdx.y * 64;
  v8f acc[4][4] = {};
  mma64(t16 + (size_t)m0 * Cc, Cc, wq + (size_t)o0 * Cc, Cc, Cc, acc);
  int lane = threadIdx.x & 31, col = lane & 15, r0 = (lane >> 4) << 3;
#pragma unroll
  for (int ti = 0; ti < 4; ++ti)
#pragma unroll
    for (int tj = 0; tj < 4; ++tj) {
      int o = o0 + 16 * tj + col;
      int which = o >> 8, head = (o >> 7) & 1, hd = o & 127;
      float bo = bias[o];
#pragma unroll
      for (int v = 0; v < 8; ++v) {
        int m = m0 + 16 * ti + r0 + v;
        int w = m >> 9, n = m & 511;
        float val = acc[ti][tj][v] + bo;
        size_t zi = (size_t)(w * HEADS + head);
        if (which == 0)      q16[(zi * Nn + n) * HDim + hd] = (half_t)(val * 0.08838834764831845f);
        else if (which == 1) k16[(zi * Nn + n) * HDim + hd] = (half_t)val;
        else                 vt16[(zi * HDim + hd) * Nn + n] = (half_t)val;
      }
    }
}

// scores = q @ k^T + mask[p]  per (window, head); stored f16.
__global__ void GEMM_ATTRS k_gemm_scores(const half_t* __restrict__ q16,
                                         const half_t* __restrict__ k16,
                                         const float* __restrict__ mask,
                                         half_t* __restrict__ s16) {
  int m0 = blockIdx.x * 64, o0 = blockIdx.y * 64, z = blockIdx.z;
  v8f acc[4][4] = {};
  mma64(q16 + ((size_t)z * Nn + m0) * HDim, HDim,
        k16 + ((size_t)z * Nn + o0) * HDim, HDim, HDim, acc);
  int p = (z >> 1) & 63;
  int lane = threadIdx.x & 31, col = lane & 15, r0 = (lane >> 4) << 3;
#pragma unroll
  for (int ti = 0; ti < 4; ++ti)
#pragma unroll
    for (int tj = 0; tj < 4; ++tj) {
      int cg = o0 + 16 * tj + col;
#pragma unroll
      for (int v = 0; v < 8; ++v) {
        int row = m0 + 16 * ti + r0 + v;
        float val = acc[ti][tj][v] + mask[((size_t)p * Nn + row) * Nn + cg];
        s16[((size_t)z * Nn + row) * Nn + cg] = (half_t)val;
      }
    }
}

// O = softmax(S) @ V  -> o16 (Mtok, Cc) with heads interleaved back.
__global__ void GEMM_ATTRS k_gemm_pv(const half_t* __restrict__ s16,
                                     const half_t* __restrict__ vt16,
                                     half_t* __restrict__ o16) {
  int m0 = blockIdx.x * 64, o0 = blockIdx.y * 64, z = blockIdx.z;
  v8f acc[4][4] = {};
  mma64(s16 + ((size_t)z * Nn + m0) * Nn, Nn,
        vt16 + ((size_t)z * HDim + o0) * Nn, Nn, Nn, acc);
  int w = z >> 1, head = z & 1;
  int lane = threadIdx.x & 31, col = lane & 15, r0 = (lane >> 4) << 3;
#pragma unroll
  for (int ti = 0; ti < 4; ++ti)
#pragma unroll
    for (int tj = 0; tj < 4; ++tj) {
      int hd = o0 + 16 * tj + col;
#pragma unroll
      for (int v = 0; v < 8; ++v) {
        int n = m0 + 16 * ti + r0 + v;
        o16[((size_t)w * Nn + n) * Cc + head * HDim + hd] = (half_t)acc[ti][tj][v];
      }
    }
}

// proj + window-reverse + roll(+4,+4) + residual add  -> x1 in d_out (f32).
__global__ void GEMM_ATTRS k_gemm_proj(const half_t* __restrict__ o16,
                                       const half_t* __restrict__ wp,
                                       const float* __restrict__ bias,
                                       const float* __restrict__ xin,
                                       float* __restrict__ xout) {
  int m0 = blockIdx.x * 64, o0 = blockIdx.y * 64;
  v8f acc[4][4] = {};
  mma64(o16 + (size_t)m0 * Cc, Cc, wp + (size_t)o0 * Cc, Cc, Cc, acc);
  int lane = threadIdx.x & 31, col = lane & 15, r0 = (lane >> 4) << 3;
#pragma unroll
  for (int ti = 0; ti < 4; ++ti)
#pragma unroll
    for (int tj = 0; tj < 4; ++tj) {
      int o = o0 + 16 * tj + col;
      float bo = bias[o];
#pragma unroll
      for (int v = 0; v < 8; ++v) {
        int m = m0 + 16 * ti + r0 + v;
        int w = m >> 9, n = m & 511;
        int bb = w >> 6, p = w & 63;
        int hn = p >> 3, wn = p & 7;
        int d = n >> 6, wh = (n >> 3) & 7, ww = n & 7;
        int hh = ((hn << 3) + wh + SHIFT) & 63;   // scatter back: roll(+SHIFT)
        int wc = ((wn << 3) + ww + SHIFT) & 63;
        size_t idx = ((size_t)((bb * Dd + d) * Hh + hh) * Wdim + wc) * Cc + o;
        xout[idx] = xin[idx] + acc[ti][tj][v] + bo;
      }
    }
}

// fc1 + exact GELU -> h16 (Mtok x 1024) f16
__global__ void GEMM_ATTRS k_gemm_fc1(const half_t* __restrict__ ln16,
                                      const half_t* __restrict__ w1,
                                      const float* __restrict__ bias,
                                      half_t* __restrict__ h16) {
  int m0 = blockIdx.x * 64, o0 = blockIdx.y * 64;
  v8f acc[4][4] = {};
  mma64(ln16 + (size_t)m0 * Cc, Cc, w1 + (size_t)o0 * Cc, Cc, Cc, acc);
  int lane = threadIdx.x & 31, col = lane & 15, r0 = (lane >> 4) << 3;
#pragma unroll
  for (int ti = 0; ti < 4; ++ti)
#pragma unroll
    for (int tj = 0; tj < 4; ++tj) {
      int o = o0 + 16 * tj + col;
      float bo = bias[o];
#pragma unroll
      for (int v = 0; v < 8; ++v) {
        int m = m0 + 16 * ti + r0 + v;
        float val = acc[ti][tj][v] + bo;
        val = 0.5f * val * (1.0f + erff(val * 0.70710678118654752f));
        h16[(size_t)m * HIDd + o] = (half_t)val;
      }
    }
}

// fc2 + residual into d_out (d_out currently holds x1).
__global__ void GEMM_ATTRS k_gemm_fc2(const half_t* __restrict__ h16,
                                      const half_t* __restrict__ w2,
                                      const float* __restrict__ bias,
                                      float* __restrict__ out) {
  int m0 = blockIdx.x * 64, o0 = blockIdx.y * 64;
  v8f acc[4][4] = {};
  mma64(h16 + (size_t)m0 * HIDd, HIDd, w2 + (size_t)o0 * HIDd, HIDd, HIDd, acc);
  int lane = threadIdx.x & 31, col = lane & 15, r0 = (lane >> 4) << 3;
#pragma unroll
  for (int ti = 0; ti < 4; ++ti)
#pragma unroll
    for (int tj = 0; tj < 4; ++tj) {
      int o = o0 + 16 * tj + col;
      float bo = bias[o];
#pragma unroll
      for (int v = 0; v < 8; ++v) {
        int m = m0 + 16 * ti + r0 + v;
        size_t idx = (size_t)m * Cc + o;
        out[idx] = out[idx] + acc[ti][tj][v] + bo;   // x1 + fc2(...)
      }
    }
}

// ---------------------------------------------------------------------------

extern "C" void kernel_launch(void* const* d_in, const int* in_sizes, int n_in,
                              void* d_out, int out_size, void* d_ws, size_t ws_size,
                              hipStream_t stream) {
  (void)in_sizes; (void)n_in; (void)out_size; (void)ws_size;
  const float* x     = (const float*)d_in[0];
  // d_in[1] (attn_mask) is unused: reference takes the shifted path.
  const float* maskS = (const float*)d_in[2];
  const float* n3g = (const float*)d_in[3];
  const float* n3b = (const float*)d_in[4];
  const float* qkvw = (const float*)d_in[5];
  const float* qkvb = (const float*)d_in[6];
  const float* projw = (const float*)d_in[7];
  const float* projb = (const float*)d_in[8];
  const float* n4g = (const float*)d_in[9];
  const float* n4b = (const float*)d_in[10];
  const float* f1w = (const float*)d_in[11];
  const float* f1b = (const float*)d_in[12];
  const float* f2w = (const float*)d_in[13];
  const float* f2b = (const float*)d_in[14];
  float* out = (float*)d_out;

  // workspace partition (256B-aligned slices, f16 buffers)
  size_t off = 0;
  auto alloc = [&](size_t nh) {
    half_t* p = (half_t*)((char*)d_ws + off);
    off += ((nh * sizeof(half_t)) + 255) & ~(size_t)255;
    return p;
  };
  half_t* wq  = alloc((size_t)768 * 256);
  half_t* wp  = alloc((size_t)256 * 256);
  half_t* w1  = alloc((size_t)1024 * 256);
  half_t* w2  = alloc((size_t)256 * 1024);
  half_t* t16 = alloc((size_t)Mtok * Cc);
  half_t* q16 = alloc((size_t)BPw * HEADS * Nn * HDim);
  half_t* k16 = alloc((size_t)BPw * HEADS * Nn * HDim);
  half_t* vt16 = alloc((size_t)BPw * HEADS * HDim * Nn);
  half_t* s16 = alloc((size_t)BPw * HEADS * Nn * Nn);
  half_t* o16 = alloc((size_t)Mtok * Cc);
  half_t* ln16 = alloc((size_t)Mtok * Cc);
  half_t* h16 = alloc((size_t)Mtok * HIDd);

  // 1) weights -> f16
  k_cast_f16<<<(768 * 256 + 255) / 256, 256, 0, stream>>>(qkvw, wq, 768 * 256);
  k_cast_f16<<<(256 * 256 + 255) / 256, 256, 0, stream>>>(projw, wp, 256 * 256);
  k_cast_f16<<<(1024 * 256 + 255) / 256, 256, 0, stream>>>(f1w, w1, 1024 * 256);
  k_cast_f16<<<(256 * 1024 + 255) / 256, 256, 0, stream>>>(f2w, w2, 256 * 1024);

  // 2) LN(norm3) + shift + window partition
  k_ln_shift_window<<<Mtok, 256, 0, stream>>>(x, n3g, n3b, t16);

  // 3) QKV projection (M=65536, K=256, O=768)
  k_gemm_qkv<<<dim3(Mtok / 64, 768 / 64), 32, 0, stream>>>(t16, wq, qkvb, q16, k16, vt16);

  // 4) scores = qs @ k^T + mask   (per window*head: 512x512, K=128)
  k_gemm_scores<<<dim3(Nn / 64, Nn / 64, BPw * HEADS), 32, 0, stream>>>(q16, k16, maskS, s16);

  // 5) softmax rows
  k_softmax<<<BPw * HEADS * Nn, 256, 0, stream>>>(s16);

  // 6) O = P @ V   (per window*head: 512x128, K=512)
  k_gemm_pv<<<dim3(Nn / 64, HDim / 64, BPw * HEADS), 32, 0, stream>>>(s16, vt16, o16);

  // 7) proj + window-reverse + unshift + residual  -> x1 in d_out
  k_gemm_proj<<<dim3(Mtok / 64, Cc / 64), 32, 0, stream>>>(o16, wp, projb, x, out);

  // 8) LN(norm4)
  k_ln_rows<<<Mtok, 256, 0, stream>>>(out, n4g, n4b, ln16);

  // 9) fc1 + GELU
  k_gemm_fc1<<<dim3(Mtok / 64, HIDd / 64), 32, 0, stream>>>(ln16, w1, f1b, h16);

  // 10) fc2 + residual into d_out
  k_gemm_fc2<<<dim3(Mtok / 64, Cc / 64), 32, 0, stream>>>(h16, w2, f2b, out);
}
